// LadderEvidenceNet_87531433492884
// MI455X (gfx1250) — matrix operand order repaired and checked by reading, not compiled, not run on hardware
//
#include <hip/hip_runtime.h>
#include <hip/hip_bf16.h>
#include <math.h>

#define DEV __device__ __forceinline__

constexpr int BB = 128, RR = 512, DD = 256;
constexpr float BETA  = 6.0f;
constexpr float LOG2E = 1.44269504088896340736f;

typedef __attribute__((ext_vector_type(16))) _Float16 v16h;
typedef __attribute__((ext_vector_type(8)))  float    v8f;

// Async copy to LDS (gfx1250 GLOBAL_LOAD_ASYNC_TO_LDS_*, tracked by ASYNCcnt)
#if defined(__AMDGCN__) && __has_builtin(__builtin_amdgcn_global_load_async_to_lds_b128) && \
    __has_builtin(__builtin_amdgcn_s_wait_asynccnt)
#define USE_ASYNC_LDS 1
#define AS1 __attribute__((address_space(1)))
#define AS3 __attribute__((address_space(3)))
typedef int v4i __attribute__((vector_size(16)));   // matches builtin param type
#endif

DEV float softplus_f(float x) {
  return fmaxf(x, 0.0f) + log1pf(expf(-fabsf(x)));
}

DEV float fast_exp2(float x) {
#if __has_builtin(__builtin_amdgcn_exp2f)
  return __builtin_amdgcn_exp2f(x);
#else
  return exp2f(x);
#endif
}

DEV float fast_rcp(float x) {
#if __has_builtin(__builtin_amdgcn_rcpf)
  return __builtin_amdgcn_rcpf(x);
#else
  return 1.0f / x;
#endif
}

// sigmoid(y/log2e): argument pre-scaled by log2(e) so exp lowers to v_exp_f32
DEV float sigm2(float y) {
  return fast_rcp(1.0f + fast_exp2(-y));
}

// ---------------------------------------------------------------------------
// Kernel 1: per-(r,d) precomputation, written transposed as [d][r] for
// coalesced access in the main kernel.
//   TH4[d*R+r] = (th0, th1, th2, th3)            (ladder thresholds)
//   AME[d*R+r] = (kappa*tanh(ineq)*log2e, me)    (slope, masked evidence wt)
// ---------------------------------------------------------------------------
__global__ __launch_bounds__(256) void ladder_prep(
    const float* __restrict__ th0,
    const float* __restrict__ delta,
    const float* __restrict__ ineq,
    const float* __restrict__ esign,
    const float* __restrict__ mask,
    const float* __restrict__ log_kappa,
    float4* __restrict__ TH4,
    float2* __restrict__ AME) {
  int idx = blockIdx.x * blockDim.x + threadIdx.x;   // idx = r*D + d
  if (idx >= RR * DD) return;
  int d = idx & (DD - 1);
  int r = idx >> 8;                                  // DD == 256

  float kappa = fminf(fmaxf(expf(log_kappa[0]), 0.5f), 50.0f);

  float t0 = th0[idx];
  float i0 = softplus_f(delta[idx * 3 + 0]);
  float i1 = softplus_f(delta[idx * 3 + 1]);
  float i2 = softplus_f(delta[idx * 3 + 2]);
  float t1 = t0 + i0;
  float t2 = t1 + i1;
  float t3 = t2 + i2;

  float a  = kappa * tanhf(ineq[idx]) * LOG2E;
  float me = sigm2(mask[idx] * LOG2E) * tanhf(esign[idx]);

  int o = d * RR + r;
  TH4[o] = make_float4(t0, t1, t2, t3);
  AME[o] = make_float2(a, me);
}

// ---------------------------------------------------------------------------
// Kernel 2: main evidence kernel. Block = 16b x 16r tile (256 threads).
// Each thread owns one (b, r) pair and loops over all d. Lanes sharing r
// across the 16 b's broadcast-read the same AME/TH4 values (16x L2 reuse).
// x-tile staged in LDS via async-to-LDS DMA when available.
// ---------------------------------------------------------------------------
__global__ __launch_bounds__(256) void ladder_main(
    const float*  __restrict__ x,
    const float*  __restrict__ lwp,      // level_w_param (4)
    const float*  __restrict__ tvec,     // t (R)
    const float4* __restrict__ TH4,
    const float2* __restrict__ AME,
    float* __restrict__ ZB) {
  __shared__ float sx[16 * DD];          // 16 KB x-tile

  int tid   = threadIdx.x;
  int rblk  = blockIdx.x & 31;           // R/16 = 32 r-tiles
  int bblk  = blockIdx.x >> 5;           // B/16 =  8 b-tiles
  int rbase = rblk * 16;
  int bbase = bblk * 16;

  // x rows [bbase, bbase+16) are one contiguous 16 KB block.
  const float* xtile = x + (size_t)bbase * DD;
#if defined(USE_ASYNC_LDS)
#pragma unroll
  for (int i = 0; i < 4; ++i) {
    int e = (i * 256 + tid) * 4;         // float index, 16B granules
    __builtin_amdgcn_global_load_async_to_lds_b128(
        (AS1 v4i*)(xtile + e),
        (AS3 v4i*)(&sx[e]), /*offset=*/0, /*cpol=*/0);
  }
  __builtin_amdgcn_s_wait_asynccnt(0);
#else
#pragma unroll
  for (int i = 0; i < 16; ++i)
    sx[i * DD + tid] = xtile[i * DD + tid];
#endif
  __syncthreads();

  int rl = tid & 15;
  int bl = tid >> 4;
  int r  = rbase + rl;

  // level weights: w_l = softplus(p)+1e-6, folded with the 2/sum(w) factor
  float w0 = softplus_f(lwp[0]) + 1e-6f;
  float w1 = softplus_f(lwp[1]) + 1e-6f;
  float w2 = softplus_f(lwp[2]) + 1e-6f;
  float w3 = softplus_f(lwp[3]) + 1e-6f;
  float ws = 2.0f / (w0 + w1 + w2 + w3);
  w0 *= ws; w1 *= ws; w2 *= ws; w3 *= ws;

  const float* xs = &sx[bl * DD];
  float acc = 0.0f;    // sum_d me * (2/W) * sum_l w_l * sigma_l
  float accm = 0.0f;   // sum_d me

#pragma unroll 4
  for (int d = 0; d < DD; ++d) {
    float2 am = AME[d * RR + r];
    float4 th = TH4[d * RR + r];
    float  xv = xs[d];
    float  a  = am.x;
    float  s  = w0 * sigm2(a * (xv - th.x))
              + w1 * sigm2(a * (xv - th.y))
              + w2 * sigm2(a * (xv - th.z))
              + w3 * sigm2(a * (xv - th.w));
    acc  = fmaf(am.y, s, acc);
    accm += am.y;
    // speculative prefetch ~16 rows ahead, once per unrolled group
    if ((d & 3) == 0)
      __builtin_prefetch(&TH4[((d + 16) & (DD - 1)) * RR + r], 0, 1);
  }

  float ev = acc - accm;                           // sum_d (2*sev-1)*me
  float z  = sigm2(BETA * LOG2E * (ev - tvec[r]));
  ZB[(bbase + bl) * RR + r] = z;
}

// ---------------------------------------------------------------------------
// Kernel 3: head projection y[b] = sum_r z[b,r]*head_w[r] + head_b via
// v_wmma_f32_16x16x32_f16. One block, 8 waves; wave w owns b-rows
// [16w, 16w+16). head_w is broadcast across the 16 N columns of the B
// fragment; the result is read from column N=0 of the C/D tile.
// Fragment packing follows ISA 7.12.2 (16-bit A 16x32 / B 32x16 layouts).
// ---------------------------------------------------------------------------
__global__ __launch_bounds__(256) void ladder_head(
    const float* __restrict__ ZB,
    const float* __restrict__ head_w,
    const float* __restrict__ head_b,
    float* __restrict__ out) {
  int lane = threadIdx.x & 31;
  int wave = threadIdx.x >> 5;
  int b0   = wave * 16;
  int m    = lane & 15;        // row within 16x16 tile
  int hi   = lane >> 4;        // half-wave selector

  v8f c = {};
  const float* zr = ZB + (size_t)(b0 + m) * RR;

  for (int kc = 0; kc < RR; kc += 32) {
    v16h a, bf;
    // A (16x32 f16): lanes 0-15 hold K {0..7,16..23}; lanes 16-31 hold
    // K {8..15,24..31} (per ISA 16-bit A-matrix table).
#pragma unroll
    for (int i = 0; i < 8; ++i) {
      a[i]     = (_Float16)zr[kc + hi * 8 + i];
      a[8 + i] = (_Float16)zr[kc + 16 + hi * 8 + i];
    }
    // B (32x16 f16): lanes 0-15 hold K {0..7,16..23}; lanes 16-31 hold
    // K {8..15,24..31}; value broadcast across all N columns.
#pragma unroll
    for (int i = 0; i < 8; ++i) {
      bf[i]     = (_Float16)head_w[kc + hi * 8 + i];
      bf[8 + i] = (_Float16)head_w[kc + 16 + hi * 8 + i];
    }

    c = __builtin_amdgcn_wmma_f32_16x16x32_f16(
        /*neg_a=*/false, a, /*neg_b=*/false, bf,
        /*c_mod=*/(short)0, c, /*reuse_a=*/false, /*reuse_b=*/false);
  }

  float bb = head_b[0];
  // C/D layout: VGPR v holds M=v (lanes 0-15) / M=8+v (lanes 16-31), N=lane%16.
  if (lane == 0) {
#pragma unroll
    for (int v = 0; v < 8; ++v) out[b0 + v] = c[v] + bb;
  } else if (lane == 16) {
#pragma unroll
    for (int v = 0; v < 8; ++v) out[b0 + 8 + v] = c[v] + bb;
  }
}

// ---------------------------------------------------------------------------
extern "C" void kernel_launch(void* const* d_in, const int* in_sizes, int n_in,
                              void* d_out, int out_size, void* d_ws, size_t ws_size,
                              hipStream_t stream) {
  const float* x          = (const float*)d_in[0];
  const float* th0        = (const float*)d_in[1];
  const float* delta      = (const float*)d_in[2];
  const float* ineq       = (const float*)d_in[3];
  const float* esign      = (const float*)d_in[4];
  const float* mask       = (const float*)d_in[5];
  const float* log_kappa  = (const float*)d_in[6];
  const float* lwp        = (const float*)d_in[7];
  const float* tvec       = (const float*)d_in[8];
  const float* head_w     = (const float*)d_in[9];
  const float* head_b     = (const float*)d_in[10];

  // workspace layout
  char* ws = (char*)d_ws;
  float4* TH4 = (float4*)ws;                                   // R*D*16 = 2 MB
  float2* AME = (float2*)(ws + (size_t)RR * DD * 16);          // R*D*8  = 1 MB
  float*  ZB  = (float*) (ws + (size_t)RR * DD * 24);          // B*R*4  = 256 KB

  ladder_prep<<<(RR * DD) / 256, 256, 0, stream>>>(
      th0, delta, ineq, esign, mask, log_kappa, TH4, AME);

  ladder_main<<<(BB / 16) * (RR / 16), 256, 0, stream>>>(
      x, lwp, tvec, TH4, AME, ZB);

  ladder_head<<<1, 256, 0, stream>>>(ZB, head_w, head_b, (float*)d_out);
}
